// MyPVT_44117904065214
// MI455X (gfx1250) — matrix operand order repaired and checked by reading, not compiled, not run on hardware
//
#include <hip/hip_runtime.h>
#include <math.h>

// ---------------- constants ----------------
constexpr int B_    = 16;
constexpr int Hm_   = 56;
constexpr int Wm_   = 56;
constexpr int C_    = 256;
constexpr int HEADS_= 4;
constexpr int D_    = 64;
constexpr int MLP_  = 1024;
constexpr int N_    = Hm_ * Wm_;          // 3136
constexpr int NGRID_= 196;                // (56/4)^2
constexpr int KDOWN_= N_ / 4;             // 784
constexpr int KADA_ = KDOWN_ - NGRID_;    // 588
constexpr int NREM_ = N_ - NGRID_;        // 2940
constexpr int H2_   = 28;
constexpr int W2_   = 28;
constexpr int POOLN_= 49;                 // 7*7 pooled tokens
constexpr int BT_   = B_ * N_;            // 50176 token rows

typedef __attribute__((ext_vector_type(16))) __bf16 v16bf;
typedef __attribute__((ext_vector_type(8)))  __bf16 v8bf;
typedef __attribute__((ext_vector_type(8)))  float  v8f;

// ---------------- helpers ----------------
__device__ __forceinline__ float waveReduceSum(float v) {
  #pragma unroll
  for (int off = 16; off > 0; off >>= 1) v += __shfl_xor(v, off, 32);
  return v;
}
__device__ __forceinline__ float geluf(float x) {
  return 0.5f * x * (1.0f + erff(x * 0.70710678118654752f));
}
__device__ __forceinline__ float hash_u01(unsigned x) {
  x ^= x >> 17; x *= 0xed5ad4bbu; x ^= x >> 11; x *= 0xac4c1b51u;
  x ^= x >> 15; x *= 0x31848babu; x ^= x >> 14;
  return (x >> 8) * (1.0f / 16777216.0f);
}

// ---------------- order build ----------------
__global__ void build_order_kernel(int* __restrict__ order) {
  int p = blockIdx.x * 256 + threadIdx.x;
  if (p >= N_) return;
  int y = p / Wm_, x = p - y * Wm_;
  int slot;
  if ((y & 3) == 2 && (x & 3) == 2) {
    slot = ((y - 2) >> 2) * 14 + ((x - 2) >> 2);
  } else {
    int nr = (y <= 2) ? 0 : ((y - 3) / 4 + 1);        // full grid rows before y
    int gb = nr * 14;
    if ((y & 3) == 2) gb += (x <= 2) ? 0 : ((x - 3) / 4 + 1);
    slot = NGRID_ + (p - gb);
  }
  order[slot] = p;
}

// ---------------- fused gather + LayerNorm (+opt GELU) ----------------
__global__ void __launch_bounds__(256) ln_kernel(
    const float* __restrict__ in, const int* __restrict__ order,
    const float* __restrict__ gamma, const float* __restrict__ beta,
    float* __restrict__ perm_out, float* __restrict__ norm_out,
    int rows, int tokens_per_batch, int act)
{
  int wave = threadIdx.x >> 5;
  int lane = threadIdx.x & 31;
  int row  = blockIdx.x * 8 + wave;
  if (row >= rows) return;
  int src = row;
  if (order) {
    int b = row / tokens_per_batch;
    int i = row - b * tokens_per_batch;
    src = b * tokens_per_batch + order[i];
  }
  const float* ip = in + (size_t)src * C_;
  float v[8];
  float s = 0.f;
  #pragma unroll
  for (int j = 0; j < 8; ++j) { v[j] = ip[j * 32 + lane]; s += v[j]; }
  s = waveReduceSum(s);
  float mean = s * (1.f / 256.f);
  float var = 0.f;
  #pragma unroll
  for (int j = 0; j < 8; ++j) { float d = v[j] - mean; var += d * d; }
  var = waveReduceSum(var) * (1.f / 256.f);
  float rstd = rsqrtf(var + 1e-5f);
  float* pp = perm_out ? perm_out + (size_t)row * C_ : nullptr;
  float* np = norm_out + (size_t)row * C_;
  #pragma unroll
  for (int j = 0; j < 8; ++j) {
    int c = j * 32 + lane;
    if (pp) pp[c] = v[j];
    float o = (v[j] - mean) * rstd * gamma[c] + beta[c];
    if (act == 2) o = geluf(o);
    np[c] = o;
  }
}

// ---------------- WMMA bf16 GEMM: out = act(A @ W + bias) + resid ----------
// A: [M,K] f32, W: [K,Nc] f32. Block tile 64x64, K-chunk 32, 8 waves;
// each wave owns one 16-row band and two 16-col subtiles (2 WMMA / chunk).
// A tile staged raw-fp32 via ASYNCcnt-tracked global_load_async_to_lds_b128
// (no VGPR round-trip; overlaps with B staging), converted to bf16 at
// fragment-read time. B tile staged transposed as bf16 so each fragment run
// is contiguous; pitches keep every run 16B-aligned -> ds_load_b128.
__global__ void __launch_bounds__(256) gemm_wmma_kernel(
    const float* __restrict__ A, const float* __restrict__ W,
    const float* __restrict__ bias, const float* __restrict__ resid,
    float* __restrict__ Cout, int M, int K, int Nc, int act)
{
  __shared__ __align__(16) float  As32[64][36];  // [m][k] fp32, pitch 144B
  __shared__ __align__(16) __bf16 Bt[64][40];    // [n][k] bf16, transposed
  int tid  = threadIdx.x;
  int wave = tid >> 5, lane = tid & 31;
  int m0 = blockIdx.x * 64;
  int n0 = blockIdx.y * 64;
  int wm = (wave >> 1) * 16;
  int wn = (wave & 1) * 32;
  int lm = lane & 15, grp = lane >> 4;

  // loader thread mappings (branchless; OOB rows clamped, results discarded)
  int ar  = tid >> 2;                 // 0..63  tile row
  int ac  = (tid & 3) * 8;            // 0,8,16,24
  int agr = min(m0 + ar, M - 1);
  int bn  = tid & 63;                 // 0..63  tile col (n)
  int bk  = (tid >> 6) * 8;           // 0,8,16,24

  unsigned lds_a = (unsigned)(size_t)&As32[ar][ac];   // LDS byte offset

  v8f acc0 = {}, acc1 = {};
  for (int k0 = 0; k0 < K; k0 += 32) {
    // ---- A tile: async DMA global->LDS, 2 x 16B per thread ----
    {
      unsigned long long g0 =
          (unsigned long long)(size_t)(A + (size_t)agr * K + k0 + ac);
      asm volatile("global_load_async_to_lds_b128 %0, %1, off"
                   :: "v"(lds_a), "v"(g0) : "memory");
      asm volatile("global_load_async_to_lds_b128 %0, %1, off"
                   :: "v"(lds_a + 16u), "v"(g0 + 16ull) : "memory");
    }
    // ---- B tile: columnar reads (coalesced across lanes for each j),
    //      bf16-convert, transposed -> one ds_store_b128 per thread ----
    v8bf bv;
    #pragma unroll
    for (int j = 0; j < 8; ++j)
      bv[j] = (__bf16)W[(size_t)(k0 + bk + j) * Nc + n0 + bn];
    *(v8bf*)&Bt[bn][bk] = bv;
    if (k0 + 32 < K)   // gfx1250 global_prefetch_b8 for next W chunk
      __builtin_prefetch(W + (size_t)(k0 + 32 + bk) * Nc + n0 + bn, 0, 1);
    // async A copies must land before the barrier publishes LDS
    asm volatile("s_wait_asynccnt 0x0" ::: "memory");
    __syncthreads();
    // ---- fragments ----
    const float4* arp = (const float4*)&As32[wm + lm][grp * 8];
    float4 f0 = arp[0], f1 = arp[1];
    const float4* arp2 = (const float4*)&As32[wm + lm][16 + grp * 8];
    float4 f2 = arp2[0], f3 = arp2[1];
    v16bf af;
    af[0]  = (__bf16)f0.x; af[1]  = (__bf16)f0.y;
    af[2]  = (__bf16)f0.z; af[3]  = (__bf16)f0.w;
    af[4]  = (__bf16)f1.x; af[5]  = (__bf16)f1.y;
    af[6]  = (__bf16)f1.z; af[7]  = (__bf16)f1.w;
    af[8]  = (__bf16)f2.x; af[9]  = (__bf16)f2.y;
    af[10] = (__bf16)f2.z; af[11] = (__bf16)f2.w;
    af[12] = (__bf16)f3.x; af[13] = (__bf16)f3.y;
    af[14] = (__bf16)f3.z; af[15] = (__bf16)f3.w;
    v8bf b0l = *(const v8bf*)&Bt[wn + lm][grp * 8];
    v8bf b0h = *(const v8bf*)&Bt[wn + lm][16 + grp * 8];
    v16bf bf0 = __builtin_shufflevector(b0l, b0h,
        0,1,2,3,4,5,6,7,8,9,10,11,12,13,14,15);
    v8bf b1l = *(const v8bf*)&Bt[wn + 16 + lm][grp * 8];
    v8bf b1h = *(const v8bf*)&Bt[wn + 16 + lm][16 + grp * 8];
    v16bf bf1 = __builtin_shufflevector(b1l, b1h,
        0,1,2,3,4,5,6,7,8,9,10,11,12,13,14,15);
    acc0 = __builtin_amdgcn_wmma_f32_16x16x32_bf16(
        false, af, false, bf0, (short)0, acc0, false, false);
    acc1 = __builtin_amdgcn_wmma_f32_16x16x32_bf16(
        false, af, false, bf1, (short)0, acc1, false, false);
    __syncthreads();
  }
  #pragma unroll
  for (int sub = 0; sub < 2; ++sub) {
    #pragma unroll
    for (int r = 0; r < 8; ++r) {
      int gr = m0 + wm + grp * 8 + r;
      int gc = n0 + wn + sub * 16 + lm;
      if (gr < M) {
        float v = sub ? acc1[r] : acc0[r];
        if (bias) v += bias[gc];
        if (act == 1) v = fmaxf(v, 0.f);
        size_t o = (size_t)gr * Nc + gc;
        if (resid) v += resid[o];
        Cout[o] = v;
      }
    }
  }
}

// ---------------- 8x8 average pool into [B][49][C] -----------------------
__global__ void pool_kernel(const float* __restrict__ xn, float* __restrict__ pooled) {
  int idx = blockIdx.x * 256 + threadIdx.x;
  if (idx >= B_ * POOLN_ * C_) return;
  int c = idx & 255;
  int m = (idx >> 8) % POOLN_;
  int b = idx / (POOLN_ * C_);
  int py = m / 7, px = m - py * 7;
  float s = 0.f;
  for (int yy = 0; yy < 8; ++yy)
    for (int xx = 0; xx < 8; ++xx) {
      int n = (py * 8 + yy) * Wm_ + (px * 8 + xx);
      s += xn[((size_t)b * N_ + n) * C_ + c];
    }
  pooled[idx] = s * (1.f / 64.f);
}

// ---------------- SR 1x1 conv (tiny): t0[bm,co] = pooled[bm,:] @ sr_w^T ----
__global__ void sr_kernel(const float* __restrict__ pooled,
                          const float* __restrict__ w, const float* __restrict__ bias,
                          float* __restrict__ t0) {
  int idx = blockIdx.x * 256 + threadIdx.x;
  if (idx >= B_ * POOLN_ * C_) return;
  int co = idx & 255;
  int bm = idx >> 8;
  const float* pp = pooled + (size_t)bm * C_;
  float s = bias[co];
  for (int ci = 0; ci < C_; ++ci) s += pp[ci] * w[co * C_ + ci];
  t0[idx] = s;
}

// ---------------- attention over 49 pooled KV tokens ----------------------
__global__ void __launch_bounds__(256) attn_kernel(
    const float* __restrict__ q, const float* __restrict__ kv,
    float* __restrict__ o)
{
  __shared__ float ks[POOLN_ * D_];
  __shared__ float vs[POOLN_ * D_];
  int bh = blockIdx.y;
  int b = bh >> 2, h = bh & 3;
  for (int i = threadIdx.x; i < POOLN_ * D_; i += 256) {
    int m = i >> 6, dd = i & 63;
    size_t base = ((size_t)(b * POOLN_ + m)) * (2 * C_) + h * D_ + dd;
    ks[i] = kv[base];
    vs[i] = kv[base + C_];
  }
  __syncthreads();
  int n = blockIdx.x * 256 + threadIdx.x;
  if (n >= N_) return;
  const float* qp = q + ((size_t)b * N_ + n) * C_ + h * D_;
  float qr[64];
  #pragma unroll
  for (int dd = 0; dd < 64; ++dd) qr[dd] = qp[dd];
  float s[POOLN_];
  float mx = -1e30f;
  for (int m = 0; m < POOLN_; ++m) {
    float a = 0.f;
    const float* kp = ks + m * D_;
    #pragma unroll 8
    for (int dd = 0; dd < 64; ++dd) a += qr[dd] * kp[dd];
    a *= 0.125f;                           // d^-0.5
    s[m] = a;
    mx = fmaxf(mx, a);
  }
  float sum = 0.f;
  for (int m = 0; m < POOLN_; ++m) { s[m] = __expf(s[m] - mx); sum += s[m]; }
  float inv = 1.f / sum;
  float* op = o + ((size_t)b * N_ + n) * C_ + h * D_;
  for (int dd = 0; dd < 64; ++dd) {
    float a = 0.f;
    for (int m = 0; m < POOLN_; ++m) a += s[m] * vs[m * D_ + dd];
    op[dd] = a * inv;
  }
}

// ---------------- depthwise 3x3 + bias + GELU -----------------------------
__global__ void dwconv_kernel(const float* __restrict__ h,
                              const float* __restrict__ w, const float* __restrict__ bias,
                              float* __restrict__ out) {
  size_t idx = (size_t)blockIdx.x * 256 + threadIdx.x;
  size_t total = (size_t)BT_ * MLP_;
  if (idx >= total) return;
  int ch = (int)(idx & (MLP_ - 1));
  size_t t = idx >> 10;            // b*N + n
  int n = (int)(t % N_);
  int b = (int)(t / N_);
  int y = n / Wm_, x = n - y * Wm_;
  float s = bias[ch];
  #pragma unroll
  for (int ky = -1; ky <= 1; ++ky) {
    int yy = y + ky;
    if (yy < 0 || yy >= Hm_) continue;
    #pragma unroll
    for (int kx = -1; kx <= 1; ++kx) {
      int xx = x + kx;
      if (xx < 0 || xx >= Wm_) continue;
      s += h[(((size_t)b * N_ + yy * Wm_ + xx) << 10) + ch] *
           w[ch * 9 + (ky + 1) * 3 + (kx + 1)];
    }
  }
  out[idx] = geluf(s);
}

// ---------------- conf head + hash-Gumbel scores --------------------------
__global__ void conf_kernel(const float* __restrict__ x2,
                            const float* __restrict__ cw, const float* __restrict__ cb,
                            float* __restrict__ scores) {
  int idx = blockIdx.x * 256 + threadIdx.x;
  if (idx >= BT_) return;
  int b = idx / N_, n = idx - b * N_;
  const float* xp = x2 + (size_t)idx * C_;
  float s = cb[0];
  for (int c = 0; c < C_; ++c) s += xp[c] * cw[c];
  if (n >= NGRID_) {
    float u = hash_u01((unsigned)(b * 2971u + n * 7919u + 12345u));
    float g = -logf(-logf(u + 1e-6f) + 1e-6f);
    scores[b * NREM_ + (n - NGRID_)] = s + g;
  }
}

// ---------------- exact top-k by rank counting ----------------------------
__global__ void __launch_bounds__(1024) topk_kernel(
    const float* __restrict__ scores, int* __restrict__ sel) {
  __shared__ float s[NREM_];
  int b = blockIdx.x;
  for (int i = threadIdx.x; i < NREM_; i += 1024) s[i] = scores[b * NREM_ + i];
  __syncthreads();
  for (int j = threadIdx.x; j < NREM_; j += 1024) {
    float sj = s[j];
    int rank = 0;
    for (int i = 0; i < NREM_; ++i) {
      float si = s[i];
      rank += (si > sj) || (si == sj && i < j);
    }
    if (rank < KADA_) sel[b * KADA_ + rank] = j;
  }
}

// ---------------- gather x_down -------------------------------------------
__global__ void gather_down_kernel(const float* __restrict__ x2,
                                   const int* __restrict__ sel,
                                   float* __restrict__ xd) {
  int idx = blockIdx.x * 256 + threadIdx.x;
  if (idx >= B_ * KDOWN_ * C_) return;
  int c = idx & 255;
  int i = (idx >> 8) % KDOWN_;
  int b = idx / (KDOWN_ * C_);
  int src = (i < NGRID_) ? i : NGRID_ + sel[b * KADA_ + (i - NGRID_)];
  xd[idx] = x2[((size_t)b * N_ + src) * C_ + c];
}

__global__ void locdown_kernel(const int* __restrict__ order,
                               const int* __restrict__ sel,
                               float* __restrict__ ld) {
  int idx = blockIdx.x * 256 + threadIdx.x;
  if (idx >= B_ * KDOWN_) return;
  int i = idx % KDOWN_, b = idx / KDOWN_;
  int src = (i < NGRID_) ? i : NGRID_ + sel[b * KADA_ + (i - NGRID_)];
  int p = order[src];
  ld[idx * 2 + 0] = 2.f * ((p % Wm_) + 0.5f) / Wm_ - 1.f;
  ld[idx * 2 + 1] = 2.f * ((p / Wm_) + 0.5f) / Hm_ - 1.f;
}

// ---------------- token2map: zero, scatter-add, normalize -----------------
__global__ void zero_kernel(float* __restrict__ p, int n) {
  int i = blockIdx.x * 256 + threadIdx.x;
  if (i < n) p[i] = 0.f;
}

__global__ void scatter_kernel(const float* __restrict__ x2,
                               const int* __restrict__ order,
                               float* __restrict__ F1) {
  int idx = blockIdx.x * 256 + threadIdx.x;
  if (idx >= B_ * N_ * (C_ + 1)) return;
  int ch = idx % (C_ + 1);
  int n  = (idx / (C_ + 1)) % N_;
  int b  = idx / ((C_ + 1) * N_);
  int p = order[n];
  float lx = 2.f * ((p % Wm_) + 0.5f) / Wm_ - 1.f;
  float ly = 2.f * ((p / Wm_) + 0.5f) / Hm_ - 1.f;
  int px = (int)roundf(0.5f * (lx + 1.f) * W2_ - 0.5f);
  int py = (int)roundf(0.5f * (ly + 1.f) * H2_ - 0.5f);
  px = min(max(px, 0), W2_ - 1);
  py = min(max(py, 0), H2_ - 1);
  int pos = py * W2_ + px;
  float v = (ch < C_) ? x2[((size_t)b * N_ + n) * C_ + ch] : 1.f;
  atomicAdd(&F1[(((size_t)b * (C_ + 1) + ch) * (H2_ * W2_)) + pos], v);
}

__global__ void normalize_kernel(float* __restrict__ F1) {
  int idx = blockIdx.x * 256 + threadIdx.x;
  if (idx >= B_ * H2_ * W2_) return;
  int pos = idx % (H2_ * W2_);
  int b   = idx / (H2_ * W2_);
  size_t base = (size_t)b * (C_ + 1) * (H2_ * W2_);
  float mask = F1[base + (size_t)C_ * (H2_ * W2_) + pos];
  float inv  = 1.f / (mask + 1e-6f);
  float mb   = (mask > 0.f) ? 1.f : 0.f;
  for (int ch = 0; ch < C_; ++ch) {
    size_t o = base + (size_t)ch * (H2_ * W2_) + pos;
    F1[o] = F1[o] * inv * mb;
  }
  F1[base + (size_t)C_ * (H2_ * W2_) + pos] = mb;
}

__global__ void gauss_kernel(const float* __restrict__ F1, float* __restrict__ F2) {
  int idx = blockIdx.x * 256 + threadIdx.x;
  if (idx >= B_ * (C_ + 1) * H2_ * W2_) return;
  int pos = idx % (H2_ * W2_);
  int bc  = idx / (H2_ * W2_);       // b*(C+1)+ch
  int y = pos / W2_, x = pos - y * W2_;
  float e1 = __expf(-0.125f), e2 = __expf(-0.25f);
  float norm = 1.f / (1.f + 4.f * e1 + 4.f * e2);
  float s = 0.f;
  #pragma unroll
  for (int ky = -1; ky <= 1; ++ky) {
    int yy = y + ky;
    if (yy < 0 || yy >= H2_) continue;
    #pragma unroll
    for (int kx = -1; kx <= 1; ++kx) {
      int xx = x + kx;
      if (xx < 0 || xx >= W2_) continue;
      int d2 = ky * ky + kx * kx;
      float w = (d2 == 0) ? 1.f : ((d2 == 1) ? e1 : e2);
      s += F1[(size_t)bc * (H2_ * W2_) + yy * W2_ + xx] * w * norm;
    }
  }
  F2[idx] = s;
}

__global__ void combine_kernel(float* __restrict__ F1, const float* __restrict__ F2) {
  int idx = blockIdx.x * 256 + threadIdx.x;
  if (idx >= B_ * C_ * H2_ * W2_) return;
  int pos = idx % (H2_ * W2_);
  int ch  = (idx / (H2_ * W2_)) % C_;
  int b   = idx / (C_ * H2_ * W2_);
  size_t base = (size_t)b * (C_ + 1) * (H2_ * W2_);
  float den = F2[base + (size_t)C_ * (H2_ * W2_) + pos];
  float fi  = F2[base + (size_t)ch * (H2_ * W2_) + pos] / (den + 1e-6f);
  fi *= (den > 0.f) ? 1.f : 0.f;
  float mb  = F1[base + (size_t)C_ * (H2_ * W2_) + pos];
  size_t o  = base + (size_t)ch * (H2_ * W2_) + pos;
  F1[o] = F1[o] + (1.f - mb) * fi;
}

// ---------------- map2token (bilinear) + residual → d_out -----------------
__global__ void map2token_kernel(const float* __restrict__ F1,
                                 const float* __restrict__ ld,
                                 const float* __restrict__ xd,
                                 float* __restrict__ out) {
  int idx = blockIdx.x * 256 + threadIdx.x;
  if (idx >= B_ * KDOWN_ * C_) return;
  int ch = idx & 255;
  int i  = (idx >> 8) % KDOWN_;
  int b  = idx / (KDOWN_ * C_);
  float lx = ld[((size_t)b * KDOWN_ + i) * 2 + 0];
  float ly = ld[((size_t)b * KDOWN_ + i) * 2 + 1];
  float gx = (lx + 1.f) * 0.5f * W2_ - 0.5f;
  float gy = (ly + 1.f) * 0.5f * H2_ - 0.5f;
  int x0 = (int)floorf(gx), y0 = (int)floorf(gy);
  float wx = gx - x0, wy = gy - y0;
  size_t base = ((size_t)b * (C_ + 1) + ch) * (H2_ * W2_);
  auto samp = [&](int xi, int yi) -> float {
    if (xi < 0 || xi >= W2_ || yi < 0 || yi >= H2_) return 0.f;
    return F1[base + yi * W2_ + xi];
  };
  float v = samp(x0, y0) * (1.f - wx) * (1.f - wy)
          + samp(x0 + 1, y0) * wx * (1.f - wy)
          + samp(x0, y0 + 1) * (1.f - wx) * wy
          + samp(x0 + 1, y0 + 1) * wx * wy;
  out[idx] = xd[idx] + v;
}

// ---------------- host launch ---------------------------------------------
extern "C" void kernel_launch(void* const* d_in, const int* in_sizes, int n_in,
                              void* d_out, int out_size, void* d_ws, size_t ws_size,
                              hipStream_t stream) {
  const float* x      = (const float*)d_in[0];
  const float* ln1_g  = (const float*)d_in[1];
  const float* ln1_b  = (const float*)d_in[2];
  const float* wq     = (const float*)d_in[3];
  const float* wkv    = (const float*)d_in[4];
  const float* wproj  = (const float*)d_in[5];
  const float* bproj  = (const float*)d_in[6];
  const float* sr_w   = (const float*)d_in[7];
  const float* sr_b   = (const float*)d_in[8];
  const float* srn_g  = (const float*)d_in[9];
  const float* srn_b  = (const float*)d_in[10];
  const float* ln2_g  = (const float*)d_in[11];
  const float* ln2_b  = (const float*)d_in[12];
  const float* fc1_w  = (const float*)d_in[13];
  const float* fc1_b  = (const float*)d_in[14];
  const float* dw_w   = (const float*)d_in[15];
  const float* dw_b   = (const float*)d_in[16];
  const float* fc2_w  = (const float*)d_in[17];
  const float* fc2_b  = (const float*)d_in[18];
  const float* conf_w = (const float*)d_in[19];
  const float* conf_b = (const float*)d_in[20];
  float* out = (float*)d_out;

  // workspace arena
  char* wp = (char*)d_ws;
  auto alloc = [&](size_t bytes) -> void* {
    void* r = (void*)wp;
    wp += (bytes + 255) & ~(size_t)255;
    return r;
  };
  int*   order  = (int*)  alloc((size_t)N_ * 4);
  float* A      = (float*)alloc((size_t)BT_ * C_ * 4);     // x_perm -> x1 -> x2
  float* Bb     = (float*)alloc((size_t)BT_ * C_ * 4);     // xn -> xn2
  float* Q      = (float*)alloc((size_t)BT_ * C_ * 4);     // q -> o
  float* H1     = (float*)alloc((size_t)BT_ * MLP_ * 4);   // fc1 out
  float* Hc     = (float*)alloc((size_t)BT_ * MLP_ * 4);   // dwconv+gelu out
  float* pooled = (float*)alloc((size_t)B_ * POOLN_ * C_ * 4);
  float* tbuf   = (float*)alloc((size_t)B_ * POOLN_ * C_ * 4);
  float* kvbuf  = (float*)alloc((size_t)B_ * POOLN_ * 2 * C_ * 4);
  float* scores = (float*)alloc((size_t)B_ * NREM_ * 4);
  int*   sel    = (int*)  alloc((size_t)B_ * KADA_ * 4);
  float* xdown  = (float*)alloc((size_t)B_ * KDOWN_ * C_ * 4);
  float* ldown  = (float*)alloc((size_t)B_ * KDOWN_ * 2 * 4);
  float* F1     = (float*)alloc((size_t)B_ * (C_ + 1) * H2_ * W2_ * 4);
  float* F2     = (float*)alloc((size_t)B_ * (C_ + 1) * H2_ * W2_ * 4);

  auto cdiv = [](long long a, long long b) { return (int)((a + b - 1) / b); };

  // 1. permutation order
  build_order_kernel<<<cdiv(N_, 256), 256, 0, stream>>>(order);
  // 2. gather + LN1
  ln_kernel<<<cdiv(BT_, 8), 256, 0, stream>>>(x, order, ln1_g, ln1_b, A, Bb, BT_, N_, 0);
  // 3. q = xn @ wq  (WMMA bf16)
  gemm_wmma_kernel<<<dim3(cdiv(BT_, 64), C_ / 64), 256, 0, stream>>>(
      Bb, wq, nullptr, nullptr, Q, BT_, C_, C_, 0);
  // 4. pool 8x8 -> [B][49][C]
  pool_kernel<<<cdiv(B_ * POOLN_ * C_, 256), 256, 0, stream>>>(Bb, pooled);
  // 5. SR 1x1 conv (tiny, sr_w is [out,in] so scalar kernel handles transpose)
  sr_kernel<<<cdiv(B_ * POOLN_ * C_, 256), 256, 0, stream>>>(pooled, sr_w, sr_b, tbuf);
  // 6. LN + GELU on pooled tokens (in place)
  ln_kernel<<<cdiv(B_ * POOLN_, 8), 256, 0, stream>>>(
      tbuf, nullptr, srn_g, srn_b, nullptr, tbuf, B_ * POOLN_, POOLN_, 2);
  // 7. kv = t @ wkv (WMMA bf16)
  gemm_wmma_kernel<<<dim3(cdiv(B_ * POOLN_, 64), (2 * C_) / 64), 256, 0, stream>>>(
      tbuf, wkv, nullptr, nullptr, kvbuf, B_ * POOLN_, C_, 2 * C_, 0);
  // 8. attention (KV cached in LDS), o overwrites q buffer
  attn_kernel<<<dim3(cdiv(N_, 256), B_ * HEADS_), 256, 0, stream>>>(Q, kvbuf, Q);
  // 9. x1 = x_perm + o @ wproj + bproj (in-place residual)
  gemm_wmma_kernel<<<dim3(cdiv(BT_, 64), C_ / 64), 256, 0, stream>>>(
      Q, wproj, bproj, A, A, BT_, C_, C_, 0);
  // 10. LN2
  ln_kernel<<<cdiv(BT_, 8), 256, 0, stream>>>(A, nullptr, ln2_g, ln2_b, nullptr, Bb, BT_, N_, 0);
  // 11. h = relu(xn2 @ fc1 + b)
  gemm_wmma_kernel<<<dim3(cdiv(BT_, 64), MLP_ / 64), 256, 0, stream>>>(
      Bb, fc1_w, fc1_b, nullptr, H1, BT_, C_, MLP_, 1);
  // 12. depthwise 3x3 + bias + gelu
  dwconv_kernel<<<cdiv((long long)BT_ * MLP_, 256), 256, 0, stream>>>(H1, dw_w, dw_b, Hc);
  // 13. x2 = x1 + h @ fc2 + b (in-place residual)
  gemm_wmma_kernel<<<dim3(cdiv(BT_, 64), C_ / 64), 256, 0, stream>>>(
      Hc, fc2_w, fc2_b, A, A, BT_, MLP_, C_, 0);
  // 14. conf + hash-Gumbel scores (JAX PRNG not reproducible on device)
  conf_kernel<<<cdiv(BT_, 256), 256, 0, stream>>>(A, conf_w, conf_b, scores);
  // 15. exact top-k via rank counting
  topk_kernel<<<B_, 1024, 0, stream>>>(scores, sel);
  // 16. gather x_down, loc_down
  gather_down_kernel<<<cdiv(B_ * KDOWN_ * C_, 256), 256, 0, stream>>>(A, sel, xdown);
  locdown_kernel<<<cdiv(B_ * KDOWN_, 256), 256, 0, stream>>>(order, sel, ldown);
  // 17. token2map: zero, scatter-add (fp32 atomics), normalize, gauss, combine
  int f1n = B_ * (C_ + 1) * H2_ * W2_;
  zero_kernel<<<cdiv(f1n, 256), 256, 0, stream>>>(F1, f1n);
  scatter_kernel<<<cdiv((long long)B_ * N_ * (C_ + 1), 256), 256, 0, stream>>>(A, order, F1);
  normalize_kernel<<<cdiv(B_ * H2_ * W2_, 256), 256, 0, stream>>>(F1);
  gauss_kernel<<<cdiv(f1n, 256), 256, 0, stream>>>(F1, F2);
  combine_kernel<<<cdiv(B_ * C_ * H2_ * W2_, 256), 256, 0, stream>>>(F1, F2);
  // 18. map2token bilinear + residual -> out
  map2token_kernel<<<cdiv(B_ * KDOWN_ * C_, 256), 256, 0, stream>>>(F1, ldown, xdown, out);
}
